// GAT_41927470744112
// MI455X (gfx1250) — matrix operand order repaired and checked by reading, not compiled, not run on hardware
//
#include <hip/hip_runtime.h>
#include <hip/hip_bf16.h>

// ---------------------------------------------------------------------------
// GAT pipeline for MI455X (gfx1250, wave32, WMMA).
// Dominant matrix work: per-edge per-head 32x32 GEMMs -> V_WMMA_F32_16X16X32_F16.
// All gathered state (x / h16, ~25MB) is L2-resident on the 192MB L2.
// ---------------------------------------------------------------------------

typedef _Float16 f16;
typedef __attribute__((ext_vector_type(16))) _Float16 v16h;
typedef __attribute__((ext_vector_type(8)))  _Float16 v8h;
typedef __attribute__((ext_vector_type(8)))  float    v8f;

static constexpr int   Nn   = 50000;
static constexpr int   Ee   = 800000;
static constexpr int   Dd   = 128;
static constexpr int   Hh   = 4;
static constexpr int   DHd  = 32;
static constexpr int   NLay = 2;
static constexpr float ALPHA = 0.3f;

// Order-preserving float<->uint key for atomicMax over signed floats.
__device__ __forceinline__ unsigned fkey(float f) {
  unsigned u = __float_as_uint(f);
  return (u & 0x80000000u) ? ~u : (u | 0x80000000u);
}
__device__ __forceinline__ float fdec(unsigned k) {
  unsigned u = (k & 0x80000000u) ? (k ^ 0x80000000u) : ~k;
  return __uint_as_float(u);
}

// -------------------------------- utility ----------------------------------
__global__ void k_zero(float* __restrict__ p, long n) {
  long i = (long)blockIdx.x * blockDim.x + threadIdx.x;
  long stride = (long)gridDim.x * blockDim.x;
  for (; i < n; i += stride) p[i] = 0.0f;
}

// ------------------------- phase A: edge_vals softmax ----------------------
// edge_vals >= 0, so raw-bit int atomicMax works with 0-init.
__global__ void k_segmax0(const float* __restrict__ vals,
                          const int* __restrict__ edges,
                          int* __restrict__ segmax, int E) {
  int e = blockIdx.x * blockDim.x + threadIdx.x;
  if (e >= E) return;
  int s = edges[2 * e];
  atomicMax(&segmax[s], __float_as_int(vals[e]));
}

__global__ void k_expsum0(const float* __restrict__ vals,
                          const int* __restrict__ edges,
                          const float* __restrict__ segmax,
                          float* __restrict__ ebuf,
                          float* __restrict__ ssum, int E) {
  int e = blockIdx.x * blockDim.x + threadIdx.x;
  if (e >= E) return;
  int s = edges[2 * e];
  float ex = __expf(vals[e] - segmax[s]);
  ebuf[e] = ex;
  atomicAdd(&ssum[s], ex);
}

// x[src,:] += (ex/ssum[src]) * emb[dst,:]   (one wave per edge, float4/lane)
__global__ void k_agg0(const float* __restrict__ emb,
                       const int* __restrict__ edges,
                       const float* __restrict__ ebuf,
                       const float* __restrict__ ssum,
                       float* __restrict__ x, int E) {
  int wv = threadIdx.x >> 5, lane = threadIdx.x & 31;
  int e = blockIdx.x * (blockDim.x >> 5) + wv;
  if (e >= E) return;
  int s = edges[2 * e], d = edges[2 * e + 1];
  float w = ebuf[e] / ssum[s];
  const float4 v = ((const float4*)(emb + (size_t)d * Dd))[lane];
  float* o = x + (size_t)s * Dd + lane * 4;
  atomicAdd(o + 0, w * v.x);
  atomicAdd(o + 1, w * v.y);
  atomicAdd(o + 2, w * v.z);
  atomicAdd(o + 3, w * v.w);
}

// ------------------------------ per-layer ----------------------------------
__global__ void k_relu_h16(const float* __restrict__ x, f16* __restrict__ h16, long n) {
  long i = (long)blockIdx.x * blockDim.x + threadIdx.x;
  long stride = (long)gridDim.x * blockDim.x;
  for (; i < n; i += stride) h16[i] = (f16)fmaxf(x[i], 0.0f);
}

// WMMA einsum + attention logits. One wave handles 16 edges.
//   A (16x32 f16): gathered head-slice of h16 for dst (neigh) / src (self)
//   B (32x16 f16): weight half-tile, pre-swizzled in LDS to the WMMA layout
//   C (16x16 f32): relu'd, dotted with attn vectors, butterfly-reduced.
__global__ void __launch_bounds__(256)
k_wmma_logits(const f16* __restrict__ h16,
              const int* __restrict__ edges,
              const float* __restrict__ Wl,      // [H][DH][DH]
              const float* __restrict__ attnl,   // [H][2*DH]
              float* __restrict__ a_buf,         // [E][H]
              unsigned* __restrict__ segmax,     // [N][H] (fkey-coded)
              unsigned* __restrict__ gmax,       // [H]    (fkey-coded)
              int ntiles) {
  __shared__ __align__(32) f16 b_lds[Hh * 2 * 32 * 16]; // 8KB, WMMA B layout
  __shared__ float attn_lds[Hh * 64];                    // 1KB

  // Stage B tiles: b_lds[((h*2+nt)*32 + lane)*16 + i] = W[h][K][f]
  //   K = (lane<16 ? 0 : 8) + (i<8 ? i : i+8),  f = nt*16 + (lane&15)
  for (int idx = threadIdx.x; idx < Hh * 2 * 32 * 16; idx += blockDim.x) {
    int i    = idx & 15;
    int lane = (idx >> 4) & 31;
    int nt   = (idx >> 9) & 1;
    int h    = idx >> 10;
    int K    = ((lane < 16) ? 0 : 8) + ((i < 8) ? i : i + 8);
    int f    = nt * 16 + (lane & 15);
    b_lds[idx] = (f16)Wl[(h * DHd + K) * DHd + f];
  }
  if (threadIdx.x < Hh * 64) attn_lds[threadIdx.x] = attnl[threadIdx.x];
  __syncthreads();

  int wv = threadIdx.x >> 5, lane = threadIdx.x & 31;
  int tile = blockIdx.x * (blockDim.x >> 5) + wv;
  if (tile >= ntiles) return;

  int m = lane & 15;
  int e = tile * 16 + m;
  int s = edges[2 * e];
  int d = edges[2 * e + 1];
  int abase = (lane < 16) ? 0 : 8; // A-layout K split across half-waves

  for (int h = 0; h < Hh; ++h) {
    // ---- A tiles (two aligned 16B loads per lane per side) ----
    const f16* rowN = h16 + ((size_t)d * Hh + h) * DHd;
    const f16* rowS = h16 + ((size_t)s * Hh + h) * DHd;
    v8h nlo = *(const v8h*)(rowN + abase);
    v8h nhi = *(const v8h*)(rowN + abase + 16);
    v8h slo = *(const v8h*)(rowS + abase);
    v8h shi = *(const v8h*)(rowS + abase + 16);
    v16h an, as;
#pragma unroll
    for (int i = 0; i < 8; ++i) {
      an[i] = nlo[i]; an[i + 8] = nhi[i];
      as[i] = slo[i]; as[i + 8] = shi[i];
    }
    // ---- B tiles from LDS (aligned 32B per lane) ----
    v16h b0 = *(const v16h*)&b_lds[((h * 2 + 0) * 32 + lane) * 16];
    v16h b1 = *(const v16h*)&b_lds[((h * 2 + 1) * 32 + lane) * 16];

    v8f cn0 = {}, cn1 = {}, cs0 = {}, cs1 = {};
    cn0 = __builtin_amdgcn_wmma_f32_16x16x32_f16(false, an, false, b0, (short)0, cn0, false, false);
    cn1 = __builtin_amdgcn_wmma_f32_16x16x32_f16(false, an, false, b1, (short)0, cn1, false, false);
    cs0 = __builtin_amdgcn_wmma_f32_16x16x32_f16(false, as, false, b0, (short)0, cs0, false, false);
    cs1 = __builtin_amdgcn_wmma_f32_16x16x32_f16(false, as, false, b1, (short)0, cs1, false, false);

    // relu(W·h) then per-lane partial of attention dot:
    //   a[m] = ws·attn[:32] + wn·attn[32:], lane holds column n = lane&15.
    const float* attnS = &attn_lds[h * 64];
    const float* attnN = attnS + 32;
    int n = lane & 15;
    float p[8];
#pragma unroll
    for (int r = 0; r < 8; ++r) {
      float vs0 = fmaxf(cs0[r], 0.0f), vs1 = fmaxf(cs1[r], 0.0f);
      float vn0 = fmaxf(cn0[r], 0.0f), vn1 = fmaxf(cn1[r], 0.0f);
      p[r] = vs0 * attnS[n] + vs1 * attnS[16 + n] + vn0 * attnN[n] + vn1 * attnN[16 + n];
    }
    // reduce over the 16 columns within each half-wave
#pragma unroll
    for (int r = 0; r < 8; ++r)
#pragma unroll
      for (int off = 8; off; off >>= 1) p[r] += __shfl_xor(p[r], off, 16);

    // redistribute so lane m holds a[m] (rows 0-7 live in lanes 0-15, 8-15 in 16-31)
    float aval = 0.0f;
#pragma unroll
    for (int r = 0; r < 8; ++r) {
      float alo = __shfl(p[r], 0, 32);
      float ahi = __shfl(p[r], 16, 32);
      if (m == r)     aval = alo;
      if (m == r + 8) aval = ahi;
    }
    aval = (aval > 0.0f) ? aval : ALPHA * aval; // leaky relu

    if (lane < 16) {
      a_buf[(size_t)e * Hh + h] = aval;
      atomicMax(&segmax[(size_t)s * Hh + h], fkey(aval));
    }
    // wave max -> one global atomic per wave per head
    float wm = aval;
#pragma unroll
    for (int off = 16; off; off >>= 1) wm = fmaxf(wm, __shfl_xor(wm, off, 32));
    if (lane == 0) atomicMax(&gmax[h], fkey(wm));
  }
}

// Global softmax denominator per head (axis=0 softmax).
__global__ void k_gsum(const float* __restrict__ a_buf,
                       const unsigned* __restrict__ gmax,
                       float* __restrict__ gsum, int E) {
  int e = blockIdx.x * blockDim.x + threadIdx.x;
  int lane = threadIdx.x & 31;
  float ex[Hh];
#pragma unroll
  for (int h = 0; h < Hh; ++h) {
    float v = (e < E) ? __expf(a_buf[(size_t)e * Hh + h] - fdec(gmax[h])) : 0.0f;
#pragma unroll
    for (int off = 16; off; off >>= 1) v += __shfl_xor(v, off, 32);
    ex[h] = v;
  }
  if (lane == 0)
#pragma unroll
    for (int h = 0; h < Hh; ++h) atomicAdd(&gsum[h], ex[h]);
}

// p = softmax_global(a); w = exp(p - segmax(p)); segment denominators.
__global__ void k_segw(const float* __restrict__ a_buf,
                       const unsigned* __restrict__ segmax,
                       const unsigned* __restrict__ gmax,
                       const float* __restrict__ gsum,
                       const int* __restrict__ edges,
                       float* __restrict__ w_buf,
                       float* __restrict__ segsum, int E) {
  int e = blockIdx.x * blockDim.x + threadIdx.x;
  if (e >= E) return;
  int s = edges[2 * e];
#pragma unroll
  for (int h = 0; h < Hh; ++h) {
    float g = fdec(gmax[h]);
    float inv = 1.0f / gsum[h];
    float p  = __expf(a_buf[(size_t)e * Hh + h] - g) * inv;
    float pm = __expf(fdec(segmax[(size_t)s * Hh + h]) - g) * inv;
    float w  = __expf(p - pm);
    w_buf[(size_t)e * Hh + h] = w;
    atomicAdd(&segsum[(size_t)s * Hh + h], w);
  }
}

// new[src,h,:] += alpha[e,h] * h16[dst,h,:]   (one wave per edge)
__global__ void k_agg(const f16* __restrict__ h16,
                      const int* __restrict__ edges,
                      const float* __restrict__ w_buf,
                      const float* __restrict__ segsum,
                      float* __restrict__ nb, int E) {
  int wv = threadIdx.x >> 5, lane = threadIdx.x & 31;
  int e = blockIdx.x * (blockDim.x >> 5) + wv;
  if (e >= E) return;
  int s = edges[2 * e], d = edges[2 * e + 1];
  int h = lane >> 3; // 8 lanes (32 floats) per head
  float alpha = w_buf[(size_t)e * Hh + h] / segsum[(size_t)s * Hh + h];
  const f16* row = h16 + (size_t)d * Dd + lane * 4;
  float* o = nb + (size_t)s * Dd + lane * 4;
  atomicAdd(o + 0, alpha * (float)row[0]);
  atomicAdd(o + 1, alpha * (float)row[1]);
  atomicAdd(o + 2, alpha * (float)row[2]);
  atomicAdd(o + 3, alpha * (float)row[3]);
}

__global__ void k_tanh_out(const float* __restrict__ nb,
                           float* __restrict__ x,
                           float* __restrict__ out, int l, long n) {
  long i = (long)blockIdx.x * blockDim.x + threadIdx.x;
  long stride = (long)gridDim.x * blockDim.x;
  for (; i < n; i += stride) {
    float v = tanhf(nb[i]);
    x[i] = v;
    long row = i / Dd, col = i % Dd;
    out[row * (NLay * Dd) + (long)l * Dd + col] = v;
  }
}

// ------------------------------- launcher ----------------------------------
extern "C" void kernel_launch(void* const* d_in, const int* in_sizes, int n_in,
                              void* d_out, int out_size, void* d_ws, size_t ws_size,
                              hipStream_t stream) {
  const float* ent_emb  = (const float*)d_in[0];
  const float* edgevals = (const float*)d_in[1];
  const float* kernels  = (const float*)d_in[2]; // [L][H][DH][DH]
  const float* attns    = (const float*)d_in[3]; // [L][H][2*DH]
  const int*   edges    = (const int*)d_in[4];   // [E][2]
  float* out = (float*)d_out;

  // workspace layout (floats); ebuf aliases w_buf, segmax0/ssum0 alias
  // segmax/segsum (re-zeroed per layer). Total ~91 MB.
  float* ws = (float*)d_ws;
  size_t o = 0;
  float* x   = ws + o; o += (size_t)Nn * Dd;
  float* nb  = ws + o; o += (size_t)Nn * Dd;
  f16*  h16  = (f16*)(ws + o); o += (size_t)Nn * Dd / 2;
  float* a_buf = ws + o; o += (size_t)Ee * Hh;
  float* w_buf = ws + o; o += (size_t)Ee * Hh;
  float* ebuf  = w_buf;
  unsigned* segmax = (unsigned*)(ws + o); o += (size_t)Nn * Hh;
  float* segsum = ws + o; o += (size_t)Nn * Hh;
  unsigned* gmax = (unsigned*)(ws + o); o += Hh;
  float* gsum = ws + o; o += Hh;

  const int ZB = 2048;
  const long ND = (long)Nn * Dd;

  // ---- phase A: edge-value segment softmax + initial aggregation ----
  k_zero<<<ZB, 256, 0, stream>>>(x, ND);
  k_zero<<<ZB, 256, 0, stream>>>((float*)segmax, (long)Nn * Hh);
  k_zero<<<ZB, 256, 0, stream>>>(segsum, (long)Nn * Hh);
  k_segmax0<<<(Ee + 255) / 256, 256, 0, stream>>>(edgevals, edges, (int*)segmax, Ee);
  k_expsum0<<<(Ee + 255) / 256, 256, 0, stream>>>(edgevals, edges, (const float*)segmax,
                                                  ebuf, segsum, Ee);
  k_agg0<<<(Ee + 7) / 8, 256, 0, stream>>>(ent_emb, edges, ebuf, segsum, x, Ee);

  // ---- layers ----
  const int ntiles = Ee / 16; // E % 16 == 0
  for (int l = 0; l < NLay; ++l) {
    k_relu_h16<<<ZB, 256, 0, stream>>>(x, h16, ND);
    k_zero<<<ZB, 256, 0, stream>>>(nb, ND);
    k_zero<<<ZB, 256, 0, stream>>>((float*)segmax, (long)Nn * Hh);
    k_zero<<<ZB, 256, 0, stream>>>(segsum, (long)Nn * Hh);
    k_zero<<<1, 32, 0, stream>>>((float*)gmax, Hh);
    k_zero<<<1, 32, 0, stream>>>(gsum, Hh);

    k_wmma_logits<<<(ntiles + 7) / 8, 256, 0, stream>>>(
        h16, edges, kernels + (size_t)l * Hh * DHd * DHd,
        attns + (size_t)l * Hh * 2 * DHd, a_buf, segmax, gmax, ntiles);
    k_gsum<<<(Ee + 255) / 256, 256, 0, stream>>>(a_buf, gmax, gsum, Ee);
    k_segw<<<(Ee + 255) / 256, 256, 0, stream>>>(a_buf, segmax, gmax, gsum,
                                                 edges, w_buf, segsum, Ee);
    k_agg<<<(Ee + 7) / 8, 256, 0, stream>>>(h16, edges, w_buf, segsum, nb, Ee);
    k_tanh_out<<<ZB, 256, 0, stream>>>(nb, x, out, l, ND);
  }
  (void)in_sizes; (void)n_in; (void)out_size; (void)ws_size;
}
